// SelfAttention_79302276153860
// MI455X (gfx1250) — matrix operand order repaired and checked by reading, not compiled, not run on hardware
//
#include <hip/hip_runtime.h>

// ---------------------------------------------------------------------------
// Self-attention, B=4 S=4096 H=1024, bf16 WMMA pipeline for gfx1250 (MI455X).
// Staged design: per-batch 64MB energy matrix lives in the 192MB L2.
// Round 2: __launch_bounds__(32,1) on GEMM kernels to unlock the full VGPR
// file for the 64x64-per-wave tile (previous round spilled accumulators).
// ---------------------------------------------------------------------------

typedef __bf16 bf16_t;
typedef __attribute__((ext_vector_type(8)))  __bf16 v8bf;
typedef __attribute__((ext_vector_type(16))) __bf16 v16bf;
typedef __attribute__((ext_vector_type(8)))  float  v8f;

constexpr int Bn = 4;
constexpr int Sn = 4096;
constexpr int Hn = 1024;
constexpr int Mn = Bn * Sn;   // 16384 rows for the QKV projection GEMM

// ---------------------------------------------------------------------------
// WMMA fragment loaders (CDNA5 ISA §7.12.2 layouts, wave32)
// ---------------------------------------------------------------------------

// A-matrix 16x32 bf16, row-major source (lda elements per row).
// lane l: row = l&15; half-wave hi = l>>4 selects K chunks
//   elements 0..7  = K in [k0 + hi*8,      +8)
//   elements 8..15 = K in [k0 + 16 + hi*8, +8)
__device__ __forceinline__ v16bf load_a_frag(const bf16_t* __restrict__ A, int lda,
                                             int m0, int k0, int lane) {
    const int l16 = lane & 15, hi = lane >> 4;
    const bf16_t* p = A + (size_t)(m0 + l16) * lda + k0 + hi * 8;
    union { v16bf v; v8bf h[2]; } u;
    u.h[0] = *(const v8bf*)(p);        // 16B aligned (k0 mult of 32, hi*8)
    u.h[1] = *(const v8bf*)(p + 16);
    return u.v;
}

// B-matrix 32x16 bf16, sourced from an N-major ("transposed") buffer Bt where
// Bt[n*ldb + k] = B[k][n]. lane l: column n = n0 + (l&15); K chunks
// [k0 + hi*8, +8) and [k0 + 16 + hi*8, +8) -> two 16B loads.
__device__ __forceinline__ v16bf load_b_frag(const bf16_t* __restrict__ Bt, int ldb,
                                             int n0, int k0, int lane) {
    const int l16 = lane & 15, hi = lane >> 4;
    const bf16_t* p = Bt + (size_t)(n0 + l16) * ldb + k0 + hi * 8;
    union { v16bf v; v8bf h[2]; } u;
    u.h[0] = *(const v8bf*)(p);
    u.h[1] = *(const v8bf*)(p + 16);
    return u.v;
}

// One wave computes a 64x64 f32 tile: 4x4 grid of 16x16 accumulators,
// K-loop in steps of 32 -> 16 v_wmma_f32_16x16x32_bf16 per step.
__device__ __forceinline__ void wave_gemm_64x64(const bf16_t* __restrict__ A, int lda,
                                                const bf16_t* __restrict__ Bt, int ldb,
                                                int m0, int n0, int klen, int lane,
                                                v8f acc[4][4]) {
    for (int k0 = 0; k0 < klen; k0 += 32) {
        v16bf a[4], b[4];
#pragma unroll
        for (int i = 0; i < 4; ++i) a[i] = load_a_frag(A, lda, m0 + 16 * i, k0, lane);
#pragma unroll
        for (int j = 0; j < 4; ++j) b[j] = load_b_frag(Bt, ldb, n0 + 16 * j, k0, lane);
#pragma unroll
        for (int i = 0; i < 4; ++i)
#pragma unroll
            for (int j = 0; j < 4; ++j)
                acc[i][j] = __builtin_amdgcn_wmma_f32_16x16x32_bf16(
                    /*neg_a=*/false, a[i], /*neg_b=*/false, b[j],
                    /*c_mod=*/(short)0, acc[i][j],
                    /*reuse_a=*/false, /*reuse_b=*/false);
    }
}

// ---------------------------------------------------------------------------
// 1) Pack x (f32 [M,H]) -> bf16, 8 elements/thread, 16B vector stores
// ---------------------------------------------------------------------------
__global__ void pack_x_kernel(const float* __restrict__ x, bf16_t* __restrict__ Xb) {
    size_t base = ((size_t)blockIdx.x * blockDim.x + threadIdx.x) * 8;
    const float4* xv = (const float4*)(x + base);
    float4 a = xv[0], c = xv[1];
    v8bf o = { (__bf16)a.x, (__bf16)a.y, (__bf16)a.z, (__bf16)a.w,
               (__bf16)c.x, (__bf16)c.y, (__bf16)c.z, (__bf16)c.w };
    *(v8bf*)(Xb + base) = o;
}

// ---------------------------------------------------------------------------
// 2) Pack weights transposed: Wt[z][d*H + h] = W_z[h*H + d] in bf16
// ---------------------------------------------------------------------------
__global__ void pack_wt_kernel(const float* __restrict__ Wq, const float* __restrict__ Wk,
                               const float* __restrict__ Wv, bf16_t* __restrict__ Wt) {
    int idx = blockIdx.x * 256 + threadIdx.x;        // 0 .. 3*H*H-1
    int z   = idx >> 20;                             // / (H*H)
    int rem = idx & (Hn * Hn - 1);
    int d   = rem >> 10;
    int h   = rem & (Hn - 1);
    const float* W = (z == 0) ? Wq : (z == 1) ? Wk : Wv;
    Wt[(size_t)idx] = (bf16_t)W[(size_t)h * Hn + d];
}

// ---------------------------------------------------------------------------
// 3) Fused QKV projection: C = Xb @ W_z + b_z.  grid.z selects Q/K/V.
//    Q is pre-scaled by 1/sqrt(H)=1/32 (folds attention scale into Q).
//    V is stored transposed per batch: Vt[b][d][s] (valid B-operand for PV).
// ---------------------------------------------------------------------------
__global__ void __launch_bounds__(32, 1)
qkv_gemm_kernel(const bf16_t* __restrict__ Xb, const bf16_t* __restrict__ Wt,
                const float* __restrict__ bq, const float* __restrict__ bk,
                const float* __restrict__ bv,
                bf16_t* __restrict__ Qb, bf16_t* __restrict__ Kb,
                bf16_t* __restrict__ Vt) {
    const int lane = threadIdx.x;
    const int l16 = lane & 15, hi = lane >> 4;
    const int m0 = blockIdx.x * 64;
    const int n0 = blockIdx.y * 64;
    const int z  = blockIdx.z;
    const bf16_t* W    = Wt + (size_t)z * Hn * Hn;
    const float*  bias = (z == 0) ? bq : (z == 1) ? bk : bv;

    v8f acc[4][4];
    const v8f vzero = {0.f, 0.f, 0.f, 0.f, 0.f, 0.f, 0.f, 0.f};
#pragma unroll
    for (int i = 0; i < 4; ++i)
#pragma unroll
        for (int j = 0; j < 4; ++j) acc[i][j] = vzero;

    wave_gemm_64x64(Xb, Hn, W, Hn, m0, n0, Hn, lane, acc);

#pragma unroll
    for (int i = 0; i < 4; ++i)
#pragma unroll
        for (int j = 0; j < 4; ++j) {
            const int col = n0 + 16 * j + l16;
            const float bcol = bias[col];
#pragma unroll
            for (int r = 0; r < 8; ++r) {
                const int row = m0 + 16 * i + r + 8 * hi;
                const float val = acc[i][j][r] + bcol;
                if (z == 0) {
                    Qb[(size_t)row * Hn + col] = (bf16_t)(val * 0.03125f); // /sqrt(1024)
                } else if (z == 1) {
                    Kb[(size_t)row * Hn + col] = (bf16_t)val;
                } else {
                    const int bb = row >> 12;           // / S
                    const int s  = row & (Sn - 1);
                    Vt[((size_t)bb * Hn + col) * Sn + s] = (bf16_t)val;
                }
            }
        }
}

// ---------------------------------------------------------------------------
// 4) Energy: E = Q_b @ K_b^T (scale already folded into Q). Per batch.
//    K_b row-major [s,d] is directly a valid B operand (column = key index,
//    contiguous d per lane).
// ---------------------------------------------------------------------------
__global__ void __launch_bounds__(32, 1)
energy_gemm_kernel(const bf16_t* __restrict__ Qb_b,
                   const bf16_t* __restrict__ Kb_b,
                   float* __restrict__ E) {
    const int lane = threadIdx.x;
    const int l16 = lane & 15, hi = lane >> 4;
    const int m0 = blockIdx.x * 64;
    const int n0 = blockIdx.y * 64;

    v8f acc[4][4];
    const v8f vzero = {0.f, 0.f, 0.f, 0.f, 0.f, 0.f, 0.f, 0.f};
#pragma unroll
    for (int i = 0; i < 4; ++i)
#pragma unroll
        for (int j = 0; j < 4; ++j) acc[i][j] = vzero;

    wave_gemm_64x64(Qb_b, Hn, Kb_b, Hn, m0, n0, Hn, lane, acc);

#pragma unroll
    for (int i = 0; i < 4; ++i)
#pragma unroll
        for (int j = 0; j < 4; ++j)
#pragma unroll
            for (int r = 0; r < 8; ++r)
                E[(size_t)(m0 + 16 * i + r + 8 * hi) * Sn + (n0 + 16 * j + l16)]
                    = acc[i][j][r];
}

// ---------------------------------------------------------------------------
// 5) Row softmax: P = softmax(E) row-wise, bf16 out. One 256-thread block/row.
// ---------------------------------------------------------------------------
__global__ void softmax_kernel(const float* __restrict__ E, bf16_t* __restrict__ Pb) {
    const int row = blockIdx.x;
    const int tid = threadIdx.x;
    const float* e = E + (size_t)row * Sn;

    float vals[16];
    float m = -3.4e38f;
#pragma unroll
    for (int t = 0; t < 16; ++t) {
        vals[t] = e[t * 256 + tid];
        m = fmaxf(m, vals[t]);
    }

    __shared__ float red[256];
    red[tid] = m;
    __syncthreads();
    for (int s = 128; s > 0; s >>= 1) {
        if (tid < s) red[tid] = fmaxf(red[tid], red[tid + s]);
        __syncthreads();
    }
    const float rowmax = red[0];
    __syncthreads();

    float sum = 0.f;
#pragma unroll
    for (int t = 0; t < 16; ++t) {
        vals[t] = __expf(vals[t] - rowmax);
        sum += vals[t];
    }
    red[tid] = sum;
    __syncthreads();
    for (int s = 128; s > 0; s >>= 1) {
        if (tid < s) red[tid] += red[tid + s];
        __syncthreads();
    }
    const float inv = 1.0f / red[0];

    bf16_t* p = Pb + (size_t)row * Sn;
#pragma unroll
    for (int t = 0; t < 16; ++t)
        p[t * 256 + tid] = (bf16_t)(vals[t] * inv);
}

// ---------------------------------------------------------------------------
// 6) PV: Out_b = P @ V_b (K-dim = 4096 keys, B operand = Vt[b][d][s]).
// ---------------------------------------------------------------------------
__global__ void __launch_bounds__(32, 1)
pv_gemm_kernel(const bf16_t* __restrict__ Pb,
               const bf16_t* __restrict__ Vt_b,
               float* __restrict__ Out_b) {
    const int lane = threadIdx.x;
    const int l16 = lane & 15, hi = lane >> 4;
    const int m0 = blockIdx.x * 64;   // query tile
    const int n0 = blockIdx.y * 64;   // d tile

    v8f acc[4][4];
    const v8f vzero = {0.f, 0.f, 0.f, 0.f, 0.f, 0.f, 0.f, 0.f};
#pragma unroll
    for (int i = 0; i < 4; ++i)
#pragma unroll
        for (int j = 0; j < 4; ++j) acc[i][j] = vzero;

    wave_gemm_64x64(Pb, Sn, Vt_b, Sn, m0, n0, Sn, lane, acc);

#pragma unroll
    for (int i = 0; i < 4; ++i)
#pragma unroll
        for (int j = 0; j < 4; ++j)
#pragma unroll
            for (int r = 0; r < 8; ++r)
                Out_b[(size_t)(m0 + 16 * i + r + 8 * hi) * Hn + (n0 + 16 * j + l16)]
                    = acc[i][j][r];
}

// ---------------------------------------------------------------------------
// Host launcher
// ---------------------------------------------------------------------------
extern "C" void kernel_launch(void* const* d_in, const int* in_sizes, int n_in,
                              void* d_out, int out_size, void* d_ws, size_t ws_size,
                              hipStream_t stream) {
    const float* x  = (const float*)d_in[0];
    const float* Wq = (const float*)d_in[1];
    const float* bq = (const float*)d_in[2];
    const float* Wk = (const float*)d_in[3];
    const float* bk = (const float*)d_in[4];
    const float* Wv = (const float*)d_in[5];
    const float* bv = (const float*)d_in[6];
    float* out = (float*)d_out;

    char* ws = (char*)d_ws;
    size_t off = 0;
    bf16_t* Xb = (bf16_t*)(ws + off); off += (size_t)Mn * Hn * 2;        //  32 MB
    bf16_t* Wt = (bf16_t*)(ws + off); off += (size_t)3 * Hn * Hn * 2;    //   6 MB
    bf16_t* Qb = (bf16_t*)(ws + off); off += (size_t)Mn * Hn * 2;        //  32 MB
    bf16_t* Kb = (bf16_t*)(ws + off); off += (size_t)Mn * Hn * 2;        //  32 MB
    bf16_t* Vt = (bf16_t*)(ws + off); off += (size_t)Mn * Hn * 2;        //  32 MB
    float*  E  = (float* )(ws + off); off += (size_t)Sn * Sn * 4;        //  64 MB (per-batch, L2-resident)
    bf16_t* Pb = (bf16_t*)(ws + off); off += (size_t)Sn * Sn * 2;        //  32 MB (per-batch)

    // 1) pack inputs to bf16
    pack_x_kernel<<<(Mn * Hn) / (256 * 8), 256, 0, stream>>>(x, Xb);
    pack_wt_kernel<<<(3 * Hn * Hn) / 256, 256, 0, stream>>>(Wq, Wk, Wv, Wt);

    // 2) fused QKV projection (z = 0:Q, 1:K, 2:V)
    qkv_gemm_kernel<<<dim3(Mn / 64, Hn / 64, 3), 32, 0, stream>>>(
        Xb, Wt, bq, bk, bv, Qb, Kb, Vt);

    // 3) per-batch attention; E/Pb regions reused so they stay hot in L2
    for (int b = 0; b < Bn; ++b) {
        energy_gemm_kernel<<<dim3(Sn / 64, Sn / 64), 32, 0, stream>>>(
            Qb + (size_t)b * Sn * Hn, Kb + (size_t)b * Sn * Hn, E);
        softmax_kernel<<<Sn, 256, 0, stream>>>(E, Pb);
        pv_gemm_kernel<<<dim3(Sn / 64, Hn / 64), 32, 0, stream>>>(
            Pb, Vt + (size_t)b * Hn * Sn, out + (size_t)b * Sn * Hn);
    }
}
